// HMultiHeadAttention_25151328485591
// MI455X (gfx1250) — matrix operand order repaired and checked by reading, not compiled, not run on hardware
//
#include <hip/hip_runtime.h>

// ---------------------------------------------------------------------------
// Hyperbolic multi-head attention for MI455X (gfx1250), wave32 + WMMA bf16.
// D = A*B + C via v_wmma_f32_16x16x32_bf16; fp32 accumulate everywhere.
// The Lorentz ambient dim (513) is split into a K=512 WMMA part plus a
// rank-1 time-component correction done in VALU.
// Attention K/V staging uses CDNA5 async memory->LDS DMA (ASYNCcnt).
// ---------------------------------------------------------------------------

typedef __bf16 bf16_t;
typedef __attribute__((ext_vector_type(16))) __bf16 v16bf;
typedef __attribute__((ext_vector_type(8)))  float  v8f;

static constexpr int Bsz = 4, Seq = 1024, Emb = 512, NH = 16, HD = 32;
static constexpr int ROWS = Bsz * Seq;            // 4096
static constexpr float RSQRT_HD = 0.17677669529663687f; // 1/sqrt(32)
static constexpr float EPS = 1e-6f;

// A-matrix (16x32 bf16) K index for VGPR pair v, lane-half h (ISA 7.12.2)
__device__ __forceinline__ int kmapA(int v, int half) {
  return ((v & 3) << 1) + ((v >> 2) << 4) + (half << 3);
}

// LDS byte offset of a generic pointer that aliases LDS (addrspacecast ->
// 32-bit local pointer -> int; backend strips the aperture base).
__device__ __forceinline__ unsigned lds_off_of(const void* p) {
  return (unsigned)(unsigned long long)
      (__attribute__((address_space(3))) const char*)(const char*)p;
}

// CDNA5 async DMA: LDS[vdst + 0] = MEM[vaddr], tracked by ASYNCcnt.
// (CDNA5 ISA; VGLOBAL operand order: vdst, vaddr, saddr)
__device__ __forceinline__ void async_ld_b32(const void* lds_dst, const void* gsrc) {
  asm volatile("global_load_async_to_lds_b32 %0, %1, off"
               :: "v"(lds_off_of(lds_dst)), "v"(gsrc)
               : "memory");
}
__device__ __forceinline__ void wait_asynccnt0() {
  asm volatile("s_wait_asynccnt 0" ::: "memory");
}

// ---------------------------------------------------------------------------
// Kernel 1: weight prep.  Wt[p][k][n] = W_p[n][k+1] (bf16), w0[p][n]=W_p[n][0],
// bc[p][n]=bias, Wot[k][n] = Wo[n][k+1], wo0[n] = Wo[n][0].
// ---------------------------------------------------------------------------
__global__ void k_prep_weights(const float* __restrict__ Wq, const float* __restrict__ Wk,
                               const float* __restrict__ Wv, const float* __restrict__ bq,
                               const float* __restrict__ bk, const float* __restrict__ bv,
                               const float* __restrict__ Wo,
                               bf16_t* __restrict__ Wt, float* __restrict__ w0,
                               float* __restrict__ bc, bf16_t* __restrict__ Wot,
                               float* __restrict__ wo0) {
  int idx = blockIdx.x * blockDim.x + threadIdx.x;
  if (idx < 3 * Emb * Emb) {
    int p = idx / (Emb * Emb), rem = idx % (Emb * Emb);
    int k = rem / Emb, n = rem % Emb;
    const float* W = (p == 0) ? Wq : (p == 1) ? Wk : Wv;   // [H*HD][513] flat
    Wt[idx] = (bf16_t)W[n * (Emb + 1) + (k + 1)];
  }
  if (idx < Emb * Emb) {
    int k = idx / Emb, n = idx % Emb;
    Wot[idx] = (bf16_t)Wo[n * (Emb + 1) + (k + 1)];
  }
  if (idx < 3 * Emb) {
    int p = idx / Emb, n = idx % Emb;
    const float* W  = (p == 0) ? Wq : (p == 1) ? Wk : Wv;
    const float* bb = (p == 0) ? bq : (p == 1) ? bk : bv;
    w0[idx] = W[n * (Emb + 1)];
    bc[idx] = bb[n];
  }
  if (idx < Emb) wo0[idx] = Wo[idx * (Emb + 1)];
}

// ---------------------------------------------------------------------------
// Kernel 2: lift prep.  One wave per row: t = sqrt(1+||x||^2), bf16 copy.
// grid = (ROWS/8, 3)
// ---------------------------------------------------------------------------
__global__ void k_prep_rows(const float* __restrict__ q, const float* __restrict__ k,
                            const float* __restrict__ v,
                            bf16_t* __restrict__ Xq, bf16_t* __restrict__ Xk,
                            bf16_t* __restrict__ Xv,
                            float* __restrict__ tq, float* __restrict__ tk,
                            float* __restrict__ tv) {
  int lane = threadIdx.x & 31, wave = threadIdx.x >> 5;
  int row = blockIdx.x * 8 + wave;
  int p = blockIdx.y;
  const float* src = (p == 0) ? q : (p == 1) ? k : v;
  bf16_t* dst = (p == 0) ? Xq : (p == 1) ? Xk : Xv;
  float* td = (p == 0) ? tq : (p == 1) ? tk : tv;
  float ss = 0.f;
#pragma unroll
  for (int u = 0; u < 16; ++u) {
    float x = src[(size_t)row * Emb + lane * 16 + u];
    dst[(size_t)row * Emb + lane * 16 + u] = (bf16_t)x;
    ss += x * x;
  }
#pragma unroll
  for (int m = 16; m >= 1; m >>= 1) ss += __shfl_xor(ss, m, 32);
  if (lane == 0) td[row] = sqrtf(1.f + ss);
}

// ---------------------------------------------------------------------------
// Kernel 3: per-head projection GEMM + Chen lift.
// For proj p, batch b: Z = X[1024x512] * Wt_p[512x512]  (WMMA bf16)
// then z += t*w0 + bias, lift per head (32 cols), write bf16 space + f32 time.
// grid = (8, 8, 12)  block = 256 (8 waves; wave = 16-row strip, 64 cols)
// ---------------------------------------------------------------------------
__global__ void __launch_bounds__(256) k_proj(
    const bf16_t* __restrict__ Xq, const bf16_t* __restrict__ Xk,
    const bf16_t* __restrict__ Xv, const float* __restrict__ tq,
    const float* __restrict__ tk, const float* __restrict__ tv,
    const bf16_t* __restrict__ Wt, const float* __restrict__ w0,
    const float* __restrict__ bc,
    bf16_t* __restrict__ QHs, bf16_t* __restrict__ KHs, bf16_t* __restrict__ VHs,
    float* __restrict__ QH0, float* __restrict__ KH0, float* __restrict__ VH0) {
  __shared__ bf16_t lA[128][34];      // A tile 128x32, pad->aligned b32 pairs
  __shared__ bf16_t lB[64][34];       // B tile stored [n][k]
  __shared__ float  lZ[8][16][68];    // per-wave f32 strips for epilogue

  int tid = threadIdx.x, lane = tid & 31, wave = tid >> 5, half = lane >> 4;
  int z = blockIdx.z, p = z >> 2, b = z & 3;
  int m0 = blockIdx.x * 128, n0 = blockIdx.y * 64;
  const bf16_t* X = ((p == 0) ? Xq : (p == 1) ? Xk : Xv) + (size_t)b * Seq * Emb;
  const float* tA = ((p == 0) ? tq : (p == 1) ? tk : tv) + (size_t)b * Seq;
  bf16_t* Hs = (p == 0) ? QHs : (p == 1) ? KHs : VHs;
  float*  H0 = (p == 0) ? QH0 : (p == 1) ? KH0 : VH0;

  v8f acc[4];
#pragma unroll
  for (int c = 0; c < 4; ++c)
#pragma unroll
    for (int r = 0; r < 8; ++r) acc[c][r] = 0.f;

  const unsigned int* Xu = (const unsigned int*)X;
  for (int kk = 0; kk < Emb; kk += 32) {
    if (kk + 32 < Emb)
      __builtin_prefetch(&X[(size_t)(m0 + (tid & 127)) * Emb + kk + 32], 0, 0);
    // stage A: 128x32 bf16 = 2048 dwords, 8 per thread
#pragma unroll
    for (int i = 0; i < 8; ++i) {
      int idx = tid + 256 * i;
      int row = idx >> 4, kp = idx & 15;
      unsigned int val = Xu[(size_t)(m0 + row) * (Emb / 2) + (kk >> 1) + kp];
      *(unsigned int*)&lA[row][kp * 2] = val;
    }
    // stage B: Wt[p][kk+k][n0+n] -> lB[n][k]
#pragma unroll
    for (int i = 0; i < 8; ++i) {
      int idx = tid + 256 * i;
      int n = idx & 63, k = idx >> 6;
      lB[n][k] = Wt[((size_t)p * Emb + kk + k) * Emb + n0 + n];
    }
    __syncthreads();
    v16bf a;
    int arow = wave * 16 + (lane & 15);
#pragma unroll
    for (int v = 0; v < 8; ++v) {
      int kf = kmapA(v, half);
      a[2 * v]     = lA[arow][kf];
      a[2 * v + 1] = lA[arow][kf + 1];
    }
#pragma unroll
    for (int c = 0; c < 4; ++c) {
      v16bf bm;
      int n = c * 16 + (lane & 15);
#pragma unroll
      for (int v = 0; v < 8; ++v) {
        int kf = kmapA(v, half);
        bm[2 * v]     = lB[n][kf];
        bm[2 * v + 1] = lB[n][kf + 1];
      }
      acc[c] = __builtin_amdgcn_wmma_f32_16x16x32_bf16(false, a, false, bm,
                                                       (short)0, acc[c], false, false);
    }
    __syncthreads();
  }
  // dump strips, then per-row lift
#pragma unroll
  for (int c = 0; c < 4; ++c)
#pragma unroll
    for (int r = 0; r < 8; ++r)
      lZ[wave][r + 8 * half][c * 16 + (lane & 15)] = acc[c][r];
  __syncthreads();

  int row = lane & 15, hh = half;          // lane -> (row, which-head-of-2)
  int s = m0 + wave * 16 + row;
  int h = (n0 >> 5) + hh;
  float t = tA[s];
  float zv[32];
  float ss = 0.f;
#pragma unroll
  for (int m = 0; m < 32; ++m) {
    int n = n0 + hh * 32 + m;
    float val = lZ[wave][row][hh * 32 + m] + t * w0[p * Emb + n] + bc[p * Emb + n];
    zv[m] = val;
    ss += val * val;
  }
  int bh = b * NH + h;
  bf16_t* o = Hs + ((size_t)bh * Seq + s) * HD;
#pragma unroll
  for (int m = 0; m < 32; ++m) o[m] = (bf16_t)zv[m];
  H0[(size_t)bh * Seq + s] = sqrtf(1.f + ss);
}

// ---------------------------------------------------------------------------
// Kernel 4: flash attention per (b,h).  K/V head-resident in LDS, staged with
// CDNA5 async memory->LDS DMA (global_load_async_to_lds_b32 / ASYNCcnt).
// scores = (2 + 2*(Qs.Ks - q0*k0))/sqrt(32); online softmax; mu = P*Vext;
// Lorentz centroid normalization; writes concat-layout space parts (f32).
// grid = (64, 8)  block = 256, dynamic LDS = 184832 B
// ---------------------------------------------------------------------------
__global__ void __launch_bounds__(256) k_attn(
    const bf16_t* __restrict__ QHs, const float* __restrict__ QH0,
    const bf16_t* __restrict__ KHs, const float* __restrict__ KH0,
    const bf16_t* __restrict__ VHs, const float* __restrict__ VH0,
    float* __restrict__ cat) {
  extern __shared__ char smem[];
  bf16_t* Ks = (bf16_t*)smem;                                   // [1024][34]
  float*  k0 = (float*)(smem + Seq * 34 * 2);                   // [1024]
  bf16_t* Vx = (bf16_t*)(smem + Seq * 34 * 2 + Seq * 4);        // [1024][50]
  bf16_t* Pb = (bf16_t*)(smem + Seq * 34 * 2 + Seq * 4 + Seq * 50 * 2); // [8][16][34]

  int tid = threadIdx.x, lane = tid & 31, wave = tid >> 5, half = lane >> 4;
  int bh = blockIdx.x, b = bh >> 4, h = bh & 15;
  const bf16_t* Kg = KHs + (size_t)bh * Seq * HD;
  const bf16_t* Vg = VHs + (size_t)bh * Seq * HD;
  const bf16_t* Qg = QHs + (size_t)bh * Seq * HD;
  const float* K0g = KH0 + (size_t)bh * Seq;
  const float* V0g = VH0 + (size_t)bh * Seq;
  const float* Q0g = QH0 + (size_t)bh * Seq;

  // --- async DMA staging: K space rows, V space rows, k0 time vector ------
  const unsigned int* Ku = (const unsigned int*)Kg;
  const unsigned int* Vu = (const unsigned int*)Vg;
  for (int i = 0; i < 64; ++i) {                 // 16384 dwords each array
    int idx = tid + 256 * i;
    int j = idx >> 4, kp = idx & 15;
    async_ld_b32(&Ks[j * 34 + kp * 2], Ku + idx);
    async_ld_b32(&Vx[j * 50 + kp * 2], Vu + idx);
  }
  for (int i = 0; i < 4; ++i) {
    int j = tid + 256 * i;
    async_ld_b32(&k0[j], K0g + j);
    Vx[j * 50 + 32] = (bf16_t)V0g[j];            // time column for centroid
#pragma unroll
    for (int c = 33; c < 50; ++c) Vx[j * 50 + c] = (bf16_t)0.f;
  }
  wait_asynccnt0();
  __syncthreads();

  int qt = blockIdx.y * 8 + wave;
  int s0 = qt * 16;

  v16bf aq;                                      // Q A-fragment (space part)
  {
    const unsigned int* Qu = (const unsigned int*)Qg;
    int arow = s0 + (lane & 15);
#pragma unroll
    for (int v = 0; v < 8; ++v) {
      int kf = kmapA(v, half);
      union { unsigned int u; bf16_t h2[2]; } cv;
      cv.u = Qu[(size_t)arow * 16 + (kf >> 1)];
      aq[2 * v] = cv.h2[0];
      aq[2 * v + 1] = cv.h2[1];
    }
  }
  float q0r[8];
#pragma unroll
  for (int r = 0; r < 8; ++r) q0r[r] = Q0g[s0 + r + 8 * half];

  float mst[8], lst[8];
  v8f O[3];
#pragma unroll
  for (int r = 0; r < 8; ++r) { mst[r] = -3.0e38f; lst[r] = 0.f; }
#pragma unroll
  for (int t = 0; t < 3; ++t)
#pragma unroll
    for (int r = 0; r < 8; ++r) O[t][r] = 0.f;

  bf16_t* pw = Pb + wave * 16 * 34;

  for (int jc = 0; jc < Seq; jc += 32) {
    float pt[2][8];
#pragma unroll
    for (int sub = 0; sub < 2; ++sub) {
      int j = jc + sub * 16 + (lane & 15);
      v16bf bk_;
#pragma unroll
      for (int v = 0; v < 8; ++v) {
        int kf = kmapA(v, half);
        bk_[2 * v]     = Ks[j * 34 + kf];
        bk_[2 * v + 1] = Ks[j * 34 + kf + 1];
      }
      v8f zacc;
#pragma unroll
      for (int r = 0; r < 8; ++r) zacc[r] = 0.f;
      v8f sc = __builtin_amdgcn_wmma_f32_16x16x32_bf16(false, aq, false, bk_,
                                                       (short)0, zacc, false, false);
      float k0j = k0[j];
#pragma unroll
      for (int r = 0; r < 8; ++r)
        pt[sub][r] = (2.f + 2.f * (sc[r] - q0r[r] * k0j)) * RSQRT_HD;
    }
    // online softmax: rows live across the 16 lanes of each half
    float mc[8];
#pragma unroll
    for (int r = 0; r < 8; ++r) mc[r] = fmaxf(pt[0][r], pt[1][r]);
#pragma unroll
    for (int m = 1; m <= 8; m <<= 1)
#pragma unroll
      for (int r = 0; r < 8; ++r) mc[r] = fmaxf(mc[r], __shfl_xor(mc[r], m, 32));
    float alpha[8];
#pragma unroll
    for (int r = 0; r < 8; ++r) {
      float mn = fmaxf(mst[r], mc[r]);
      alpha[r] = __expf(mst[r] - mn);
      mst[r] = mn;
    }
    float rs[8];
#pragma unroll
    for (int r = 0; r < 8; ++r) {
      pt[0][r] = __expf(pt[0][r] - mst[r]);
      pt[1][r] = __expf(pt[1][r] - mst[r]);
      rs[r] = pt[0][r] + pt[1][r];
    }
#pragma unroll
    for (int m = 1; m <= 8; m <<= 1)
#pragma unroll
      for (int r = 0; r < 8; ++r) rs[r] += __shfl_xor(rs[r], m, 32);
#pragma unroll
    for (int r = 0; r < 8; ++r) lst[r] = lst[r] * alpha[r] + rs[r];
#pragma unroll
    for (int t = 0; t < 3; ++t)
#pragma unroll
      for (int r = 0; r < 8; ++r) O[t][r] *= alpha[r];
    // stage P (16x32 bf16) through wave-private LDS to get A layout
#pragma unroll
    for (int sub = 0; sub < 2; ++sub)
#pragma unroll
      for (int r = 0; r < 8; ++r)
        pw[(r + 8 * half) * 34 + sub * 16 + (lane & 15)] = (bf16_t)pt[sub][r];
    asm volatile("s_wait_dscnt 0" ::: "memory");   // wave-internal LDS RAW
    v16bf pa;
#pragma unroll
    for (int v = 0; v < 8; ++v) {
      int kf = kmapA(v, half);
      pa[2 * v]     = pw[(lane & 15) * 34 + kf];
      pa[2 * v + 1] = pw[(lane & 15) * 34 + kf + 1];
    }
#pragma unroll
    for (int t = 0; t < 3; ++t) {
      int n = t * 16 + (lane & 15);
      v16bf bv_;
#pragma unroll
      for (int v = 0; v < 8; ++v) {
        int j0 = jc + kmapA(v, half);
        bv_[2 * v]     = Vx[j0 * 50 + n];
        bv_[2 * v + 1] = Vx[(j0 + 1) * 50 + n];
      }
      O[t] = __builtin_amdgcn_wmma_f32_16x16x32_bf16(false, pa, false, bv_,
                                                     (short)0, O[t], false, false);
    }
  }
  // Lorentz centroid normalization: out = mu / sqrt(clip(t^2 - |s|^2, eps))
  float ssum[8];
#pragma unroll
  for (int r = 0; r < 8; ++r) ssum[r] = O[0][r] * O[0][r] + O[1][r] * O[1][r];
#pragma unroll
  for (int m = 1; m <= 8; m <<= 1)
#pragma unroll
    for (int r = 0; r < 8; ++r) ssum[r] += __shfl_xor(ssum[r], m, 32);
  int srcLane = half * 16;                       // lane holding time col (n=32)
#pragma unroll
  for (int r = 0; r < 8; ++r) {
    float tt = __shfl(O[2][r], srcLane, 32);
    float l = lst[r];
    float inv = rsqrtf(fmaxf(tt * tt - ssum[r], l * l * EPS));
    int s = s0 + r + 8 * half;
    float* crow = cat + ((size_t)b * Seq + s) * Emb + h * HD;
    crow[(lane & 15)] = O[0][r] * inv;
    crow[16 + (lane & 15)] = O[1][r] * inv;
  }
}

// ---------------------------------------------------------------------------
// Kernel 5: concat lift prep — t_cat = sqrt(1+||cat||^2), bf16 copy.
// ---------------------------------------------------------------------------
__global__ void k_prep3(const float* __restrict__ cat, bf16_t* __restrict__ catbf,
                        float* __restrict__ tcat) {
  int lane = threadIdx.x & 31, wave = threadIdx.x >> 5;
  int row = blockIdx.x * 8 + wave;
  float ss = 0.f;
#pragma unroll
  for (int u = 0; u < 16; ++u) {
    float x = cat[(size_t)row * Emb + lane * 16 + u];
    catbf[(size_t)row * Emb + lane * 16 + u] = (bf16_t)x;
    ss += x * x;
  }
#pragma unroll
  for (int m = 16; m >= 1; m >>= 1) ss += __shfl_xor(ss, m, 32);
  if (lane == 0) tcat[row] = sqrtf(1.f + ss);
}

// ---------------------------------------------------------------------------
// Kernel 6: out projection GEMM 4096x512x512 + rank-1 time + bias.
// Writes out[...][1+n]; time component filled by k_time.
// grid = (32, 8), block 256
// ---------------------------------------------------------------------------
__global__ void __launch_bounds__(256) k_outproj(
    const bf16_t* __restrict__ A, const float* __restrict__ tcat,
    const bf16_t* __restrict__ Wot, const float* __restrict__ wo0,
    const float* __restrict__ bo, float* __restrict__ out) {
  __shared__ bf16_t lA[128][34];
  __shared__ bf16_t lB[64][34];
  int tid = threadIdx.x, lane = tid & 31, wave = tid >> 5, half = lane >> 4;
  int m0 = blockIdx.x * 128, n0 = blockIdx.y * 64;

  v8f acc[4];
#pragma unroll
  for (int c = 0; c < 4; ++c)
#pragma unroll
    for (int r = 0; r < 8; ++r) acc[c][r] = 0.f;

  const unsigned int* Au = (const unsigned int*)A;
  for (int kk = 0; kk < Emb; kk += 32) {
#pragma unroll
    for (int i = 0; i < 8; ++i) {
      int idx = tid + 256 * i;
      int row = idx >> 4, kp = idx & 15;
      unsigned int val = Au[(size_t)(m0 + row) * (Emb / 2) + (kk >> 1) + kp];
      *(unsigned int*)&lA[row][kp * 2] = val;
    }
#pragma unroll
    for (int i = 0; i < 8; ++i) {
      int idx = tid + 256 * i;
      int n = idx & 63, k = idx >> 6;
      lB[n][k] = Wot[(size_t)(kk + k) * Emb + n0 + n];
    }
    __syncthreads();
    v16bf a;
    int arow = wave * 16 + (lane & 15);
#pragma unroll
    for (int v = 0; v < 8; ++v) {
      int kf = kmapA(v, half);
      a[2 * v]     = lA[arow][kf];
      a[2 * v + 1] = lA[arow][kf + 1];
    }
#pragma unroll
    for (int c = 0; c < 4; ++c) {
      v16bf bm;
      int n = c * 16 + (lane & 15);
#pragma unroll
      for (int v = 0; v < 8; ++v) {
        int kf = kmapA(v, half);
        bm[2 * v]     = lB[n][kf];
        bm[2 * v + 1] = lB[n][kf + 1];
      }
      acc[c] = __builtin_amdgcn_wmma_f32_16x16x32_bf16(false, a, false, bm,
                                                       (short)0, acc[c], false, false);
    }
    __syncthreads();
  }
#pragma unroll
  for (int c = 0; c < 4; ++c)
#pragma unroll
    for (int r = 0; r < 8; ++r) {
      int row = m0 + wave * 16 + r + 8 * half;
      int n = n0 + c * 16 + (lane & 15);
      float val = acc[c][r] + tcat[row] * wo0[n] + bo[n];
      out[(size_t)row * (Emb + 1) + 1 + n] = val;
    }
}

// ---------------------------------------------------------------------------
// Kernel 7: final lift time component.
// ---------------------------------------------------------------------------
__global__ void k_time(float* __restrict__ out) {
  int lane = threadIdx.x & 31, wave = threadIdx.x >> 5;
  int row = blockIdx.x * 8 + wave;
  float ss = 0.f;
#pragma unroll
  for (int u = 0; u < 16; ++u) {
    float x = out[(size_t)row * (Emb + 1) + 1 + lane * 16 + u];
    ss += x * x;
  }
#pragma unroll
  for (int m = 16; m >= 1; m >>= 1) ss += __shfl_xor(ss, m, 32);
  if (lane == 0) out[(size_t)row * (Emb + 1)] = sqrtf(1.f + ss);
}

// ---------------------------------------------------------------------------
extern "C" void kernel_launch(void* const* d_in, const int* in_sizes, int n_in,
                              void* d_out, int out_size, void* d_ws, size_t ws_size,
                              hipStream_t stream) {
  (void)in_sizes; (void)n_in; (void)out_size; (void)ws_size;
  const float* q_space = (const float*)d_in[0];
  const float* k_space = (const float*)d_in[1];
  const float* v_space = (const float*)d_in[2];
  const float* Wq = (const float*)d_in[3];
  const float* bq = (const float*)d_in[4];
  const float* Wk = (const float*)d_in[5];
  const float* bk = (const float*)d_in[6];
  const float* Wv = (const float*)d_in[7];
  const float* bv = (const float*)d_in[8];
  const float* Wo = (const float*)d_in[9];
  const float* bo = (const float*)d_in[10];
  float* out = (float*)d_out;

  char* ws = (char*)d_ws;
  size_t off = 0;
  auto alloc = [&](size_t bytes) -> char* {
    char* p = ws + off;
    off = (off + bytes + 255) & ~(size_t)255;
    return p;
  };
  bf16_t* Xq = (bf16_t*)alloc((size_t)ROWS * Emb * 2);
  bf16_t* Xk = (bf16_t*)alloc((size_t)ROWS * Emb * 2);
  bf16_t* Xv = (bf16_t*)alloc((size_t)ROWS * Emb * 2);
  float* tq = (float*)alloc((size_t)ROWS * 4);
  float* tk = (float*)alloc((size_t)ROWS * 4);
  float* tv = (float*)alloc((size_t)ROWS * 4);
  bf16_t* Wt = (bf16_t*)alloc((size_t)3 * Emb * Emb * 2);
  float* w0 = (float*)alloc((size_t)3 * Emb * 4);
  float* bc = (float*)alloc((size_t)3 * Emb * 4);
  bf16_t* Wot = (bf16_t*)alloc((size_t)Emb * Emb * 2);
  float* wo0 = (float*)alloc((size_t)Emb * 4);
  bf16_t* QHs = (bf16_t*)alloc((size_t)Bsz * NH * Seq * HD * 2);
  bf16_t* KHs = (bf16_t*)alloc((size_t)Bsz * NH * Seq * HD * 2);
  bf16_t* VHs = (bf16_t*)alloc((size_t)Bsz * NH * Seq * HD * 2);
  float* QH0 = (float*)alloc((size_t)Bsz * NH * Seq * 4);
  float* KH0 = (float*)alloc((size_t)Bsz * NH * Seq * 4);
  float* VH0 = (float*)alloc((size_t)Bsz * NH * Seq * 4);
  float* cat = (float*)alloc((size_t)ROWS * Emb * 4);
  bf16_t* catbf = (bf16_t*)alloc((size_t)ROWS * Emb * 2);
  float* tcat = (float*)alloc((size_t)ROWS * 4);

  constexpr unsigned SMEM_ATTN = Seq * 34 * 2 + Seq * 4 + Seq * 50 * 2 + 8 * 16 * 34 * 2;
  hipFuncSetAttribute(reinterpret_cast<const void*>(k_attn),
                      hipFuncAttributeMaxDynamicSharedMemorySize, SMEM_ATTN);

  k_prep_weights<<<dim3((3 * Emb * Emb + 255) / 256), 256, 0, stream>>>(
      Wq, Wk, Wv, bq, bk, bv, Wo, Wt, w0, bc, Wot, wo0);
  k_prep_rows<<<dim3(ROWS / 8, 3), 256, 0, stream>>>(
      q_space, k_space, v_space, Xq, Xk, Xv, tq, tk, tv);
  k_proj<<<dim3(8, 8, 12), 256, 0, stream>>>(
      Xq, Xk, Xv, tq, tk, tv, Wt, w0, bc, QHs, KHs, VHs, QH0, KH0, VH0);
  k_attn<<<dim3(Bsz * NH, 8), 256, SMEM_ATTN, stream>>>(
      QHs, QH0, KHs, KH0, VHs, VH0, cat);
  k_prep3<<<dim3(ROWS / 8), 256, 0, stream>>>(cat, catbf, tcat);
  k_outproj<<<dim3(32, 8), 256, 0, stream>>>(catbf, tcat, Wot, wo0, bo, out);
  k_time<<<dim3(ROWS / 8), 256, 0, stream>>>(out);
}
